// GNNClassifier_4209067950756
// MI455X (gfx1250) — compile-verified, matrix operand order checked by reading, not run on hardware
//
#include <hip/hip_runtime.h>

// ---------------------------------------------------------------------------
// GNN classifier for MI455X (gfx1250, wave32).
// Dense GEMMs -> bf16 WMMA (v_wmma_f32_16x16x32_bf16), fp32 accumulate.
// Sparse propagation -> coalesced gather + global f32 atomics (HBM-bound).
// ---------------------------------------------------------------------------

typedef __attribute__((ext_vector_type(16))) __bf16 v16bf;
typedef __attribute__((ext_vector_type(8)))  __bf16 v8bf;
typedef __attribute__((ext_vector_type(8)))  float  v8f;

#define LEAKY   0.2f
#define G_ALPHA 0.5f
#define G_BETA  0.04879016416943205f   /* log(1 + 0.1/2) */

__device__ __forceinline__ unsigned short f2bf(float f) {
    union { float f; unsigned u; } v; v.f = f;
    unsigned r = v.u + 0x7FFFu + ((v.u >> 16) & 1u);   // round-to-nearest-even
    return (unsigned short)(r >> 16);
}

// ---------------------------------------------------------------------------
// WMMA GEMM: C[M,N] = A[M,K] @ B[K,N], fp32 in/out, bf16 multiply.
// Block: 256 threads = 8 waves (4x2), tile BM=128 x BN=64, BK=32.
// Each wave: 32x32 output = 2x2 v_wmma_f32_16x16x32_bf16 fragments.
// ---------------------------------------------------------------------------
#define BM 128
#define BN 64
#define BK 32
#define KPAD 8   // pad to 40 shorts = 80 B rows (16B-aligned, conflict-spread)

__global__ __launch_bounds__(256) void gemm_wmma_bf16(
    const float* __restrict__ A, const float* __restrict__ B,
    float* __restrict__ C, int M, int K, int N)
{
    __shared__ unsigned short As[BM][BK + KPAD];
    __shared__ unsigned short Bs[BN][BK + KPAD];

    const int t    = threadIdx.x;
    const int lane = t & 31;
    const int wave = t >> 5;
    const int wm   = wave >> 1;     // 0..3 (M dir)
    const int wn   = wave & 1;      // 0..1 (N dir)
    const int m0   = blockIdx.x * BM;
    const int n0   = blockIdx.y * BN;

    v8f acc[2][2] = {};

    for (int k0 = 0; k0 < K; k0 += BK) {
        // stage A tile (128x32 f32 -> bf16), float4 loads, coalesced along K
        #pragma unroll
        for (int j = 0; j < 4; ++j) {
            int lin = t + j * 256;           // float4 index 0..1023
            int r   = lin >> 3;              // row 0..127
            int c4  = (lin & 7) << 2;        // k 0,4,..,28
            float4 v = make_float4(0.f, 0.f, 0.f, 0.f);
            int gr = m0 + r;
            if (gr < M) v = *(const float4*)(A + (size_t)gr * K + k0 + c4);
            As[r][c4 + 0] = f2bf(v.x); As[r][c4 + 1] = f2bf(v.y);
            As[r][c4 + 2] = f2bf(v.z); As[r][c4 + 3] = f2bf(v.w);
        }
        // stage B tile transposed: Bs[n][k] (so a fragment's column is
        // contiguous in LDS, same addressing pattern as A)
        #pragma unroll
        for (int j = 0; j < 8; ++j) {
            int lin = t + j * 256;           // 0..2047
            int kk  = lin >> 6;              // 0..31
            int nn  = lin & 63;              // 0..63  (coalesced along N)
            float v = 0.f;
            if (n0 + nn < N) v = B[(size_t)(k0 + kk) * N + n0 + nn];
            Bs[nn][kk] = f2bf(v);
        }
        __syncthreads();

        // CDNA5 16-bit A/B fragment layout:
        // lanes 0-15 : row (lane), K {0..7} + {16..23}
        // lanes 16-31: row (lane-16), K {8..15} + {24..31}
        const int kc = (lane >> 4) << 3;     // 0 or 8
        #pragma unroll
        for (int sm = 0; sm < 2; ++sm) {
            int ar = wm * 32 + sm * 16 + (lane & 15);
            v8bf alo = *(const v8bf*)&As[ar][kc];
            v8bf ahi = *(const v8bf*)&As[ar][kc + 16];
            v16bf a = __builtin_shufflevector(alo, ahi,
                        0,1,2,3,4,5,6,7,8,9,10,11,12,13,14,15);
            #pragma unroll
            for (int sn = 0; sn < 2; ++sn) {
                int bc = wn * 32 + sn * 16 + (lane & 15);
                v8bf blo = *(const v8bf*)&Bs[bc][kc];
                v8bf bhi = *(const v8bf*)&Bs[bc][kc + 16];
                v16bf b = __builtin_shufflevector(blo, bhi,
                            0,1,2,3,4,5,6,7,8,9,10,11,12,13,14,15);
                acc[sm][sn] = __builtin_amdgcn_wmma_f32_16x16x32_bf16(
                    false, a, false, b, (short)0, acc[sm][sn], false, false);
            }
        }
        __syncthreads();
    }

    // C/D layout: lanes 0-15 -> N=lane, VGPR r -> M=r; lanes 16-31 -> M=r+8
    #pragma unroll
    for (int sm = 0; sm < 2; ++sm)
        #pragma unroll
        for (int sn = 0; sn < 2; ++sn) {
            int ncol  = n0 + wn * 32 + sn * 16 + (lane & 15);
            int mbase = m0 + wm * 32 + sm * 16 + ((lane >> 4) << 3);
            if (ncol < N) {
                #pragma unroll
                for (int r = 0; r < 8; ++r) {
                    int mr = mbase + r;
                    if (mr < M) C[(size_t)mr * N + ncol] = acc[sm][sn][r];
                }
            }
        }
}

// ---------------------------------------------------------------------------
// Elementwise / graph kernels
// ---------------------------------------------------------------------------
__global__ void k_fill(float* p, float v, long n) {
    long i = (long)blockIdx.x * blockDim.x + threadIdx.x;
    if (i < n) p[i] = v;
}

__global__ void k_edges(const int* __restrict__ ei, int* src, int* dst, int E, int M) {
    int i = blockIdx.x * blockDim.x + threadIdx.x;
    if (i >= M) return;
    if (i < E) { src[i] = ei[i]; dst[i] = ei[E + i]; }
    else       { src[i] = i - E; dst[i] = i - E; }
}

__global__ void k_deg(const int* __restrict__ dst, float* deg, int M) {
    int i = blockIdx.x * blockDim.x + threadIdx.x;
    if (i < M) atomicAdd(&deg[dst[i]], 1.0f);
}

__global__ void k_dinv(float* deg, int Nn) {          // in-place deg -> d^{-1/2}
    int i = blockIdx.x * blockDim.x + threadIdx.x;
    if (i < Nn) { float d = deg[i]; deg[i] = d > 0.f ? rsqrtf(d) : 0.f; }
}

__global__ void k_norm(const int* __restrict__ src, const int* __restrict__ dst,
                       const float* __restrict__ dinv, float* norm, int M) {
    int i = blockIdx.x * blockDim.x + threadIdx.x;
    if (i < M) norm[i] = dinv[src[i]] * dinv[dst[i]];
}

// out[dst] += x[src]*norm  (out pre-zeroed), float4-vectorized gather
__global__ void k_prop(const float* __restrict__ x, const int* __restrict__ src,
                       const int* __restrict__ dst, const float* __restrict__ norm,
                       float* out, int M, int F) {
    long total = (long)M * (F >> 2);
    long i = (long)blockIdx.x * blockDim.x + threadIdx.x;
    if (i >= total) return;
    int e = (int)(i / (F >> 2));
    int c = (int)(i % (F >> 2)) << 2;
    int s = src[e], d = dst[e];
    float w = norm[e];
    float4 v = *(const float4*)(x + (size_t)s * F + c);
    float* o = out + (size_t)d * F + c;
    atomicAdd(o + 0, v.x * w); atomicAdd(o + 1, v.y * w);
    atomicAdd(o + 2, v.z * w); atomicAdd(o + 3, v.w * w);
}

__global__ void k_bias_relu(const float* __restrict__ z, const float* __restrict__ b,
                            float* out, long n, int F) {
    long i = (long)blockIdx.x * blockDim.x + threadIdx.x;
    if (i < n) out[i] = fmaxf(z[i] + b[i % F], 0.f);
}

__global__ void k_combine_alpha(const float* __restrict__ p, const float* __restrict__ x0,
                                float* out, long n) {
    long i = (long)blockIdx.x * blockDim.x + threadIdx.x;
    if (i < n) out[i] = (1.f - G_ALPHA) * p[i] + G_ALPHA * x0[i];
}

__global__ void k_gcn2_out(const float* __restrict__ h, const float* __restrict__ g,
                           float* out, long n) {
    long i = (long)blockIdx.x * blockDim.x + threadIdx.x;
    if (i < n) out[i] = fmaxf((1.f - G_BETA) * h[i] + G_BETA * g[i], 0.f);
}

// ----- GATv2 -----
__global__ void k_gat_logits(const float* __restrict__ gl, const float* __restrict__ gr,
                             const float* __restrict__ att, const int* __restrict__ src,
                             const int* __restrict__ dst, float* elog, int M) {
    int i = blockIdx.x * blockDim.x + threadIdx.x;
    if (i >= M * 4) return;
    int e = i >> 2, h = i & 3;
    int s = src[e], d = dst[e];
    const float* pl = gl + (size_t)s * 128 + h * 32;
    const float* pr = gr + (size_t)d * 128 + h * 32;
    const float* pa = att + h * 32;
    float sum = 0.f;
    #pragma unroll 8
    for (int c = 0; c < 32; ++c) {
        float v = pl[c] + pr[c];
        v = v > 0.f ? v : LEAKY * v;
        sum += v * pa[c];
    }
    elog[i] = sum;
}

__device__ __forceinline__ void atomicMaxF(float* addr, float val) {
    unsigned* ua = (unsigned*)addr;
    unsigned old = *ua;
    while (__uint_as_float(old) < val) {
        unsigned assumed = old;
        old = atomicCAS(ua, assumed, __float_as_uint(val));
        if (old == assumed) break;
    }
}

__global__ void k_gat_max(const float* __restrict__ elog, const int* __restrict__ dst,
                          float* maxb, int M) {
    int i = blockIdx.x * blockDim.x + threadIdx.x;
    if (i >= M * 4) return;
    atomicMaxF(&maxb[(size_t)dst[i >> 2] * 4 + (i & 3)], elog[i]);
}

__global__ void k_gat_exp(float* elog, const int* __restrict__ dst,
                          const float* __restrict__ maxb, float* denom, int M) {
    int i = blockIdx.x * blockDim.x + threadIdx.x;
    if (i >= M * 4) return;
    int d = dst[i >> 2], h = i & 3;
    float m = maxb[(size_t)d * 4 + h];
    if (!isfinite(m)) m = 0.f;
    float ex = __expf(elog[i] - m);
    elog[i] = ex;                                   // reuse as ex
    atomicAdd(&denom[(size_t)d * 4 + h], ex);
}

__global__ void k_gat_aggr(const float* __restrict__ gl, const float* __restrict__ elog,
                           const float* __restrict__ denom, const int* __restrict__ src,
                           const int* __restrict__ dst, float* out, int M) {
    long total = (long)M * 32;                       // 128 ch / 4 per thread
    long i = (long)blockIdx.x * blockDim.x + threadIdx.x;
    if (i >= total) return;
    int e  = (int)(i >> 5);
    int cc = (int)(i & 31) << 2;                     // channel 0,4,..,124
    int h  = cc >> 5;
    int s = src[e], d = dst[e];
    float a = elog[(size_t)e * 4 + h] / (denom[(size_t)d * 4 + h] + 1e-16f);
    float4 g = *(const float4*)(gl + (size_t)s * 128 + cc);
    float* o = out + (size_t)d * 128 + cc;
    atomicAdd(o + 0, g.x * a); atomicAdd(o + 1, g.y * a);
    atomicAdd(o + 2, g.z * a); atomicAdd(o + 3, g.w * a);
}

// final: softmax over 16 classes of (z + b_cls)
__global__ void k_softmax16(const float* __restrict__ z, const float* __restrict__ b,
                            float* out, int Nn) {
    int r = blockIdx.x * blockDim.x + threadIdx.x;
    if (r >= Nn) return;
    float v[16], mx = -3.4e38f;
    #pragma unroll
    for (int c = 0; c < 16; ++c) { v[c] = z[(size_t)r * 16 + c] + b[c]; mx = fmaxf(mx, v[c]); }
    float sum = 0.f;
    #pragma unroll
    for (int c = 0; c < 16; ++c) { v[c] = __expf(v[c] - mx); sum += v[c]; }
    float inv = 1.f / sum;
    #pragma unroll
    for (int c = 0; c < 16; ++c) out[(size_t)r * 16 + c] = v[c] * inv;
}

// ---------------------------------------------------------------------------
extern "C" void kernel_launch(void* const* d_in, const int* in_sizes, int n_in,
                              void* d_out, int out_size, void* d_ws, size_t ws_size,
                              hipStream_t stream) {
    const float* x      = (const float*)d_in[0];
    const int*   ei     = (const int*)  d_in[1];
    const float* w_g1   = (const float*)d_in[2];  const float* b_g1 = (const float*)d_in[3];
    const float* w_g2   = (const float*)d_in[4];  const float* b_g2 = (const float*)d_in[5];
    const float* w_g3   = (const float*)d_in[6];  const float* b_g3 = (const float*)d_in[7];
    const float* w_g4   = (const float*)d_in[8];  const float* b_g4 = (const float*)d_in[9];
    const float* w_attl = (const float*)d_in[10]; const float* w_attr = (const float*)d_in[11];
    const float* att    = (const float*)d_in[12]; const float* b_att = (const float*)d_in[13];
    const float* w_c1   = (const float*)d_in[14];
    const float* w_g5   = (const float*)d_in[15]; const float* b_g5 = (const float*)d_in[16];
    const float* w_c2   = (const float*)d_in[17];
    const float* w_cls  = (const float*)d_in[18]; const float* b_cls = (const float*)d_in[19];
    float* out = (float*)d_out;

    const int N = in_sizes[0] / 256;
    const int E = in_sizes[1] / 2;
    const int M = E + N;

    // ---- carve workspace ----
    char* wp = (char*)d_ws;
    auto carve = [&](size_t bytes) -> void* {
        void* p = wp; wp += (bytes + 255) & ~(size_t)255; return p;
    };
    int*   srcA  = (int*)  carve((size_t)M * 4);
    int*   dstA  = (int*)  carve((size_t)M * 4);
    float* norm  = (float*)carve((size_t)M * 4);
    float* dinv  = (float*)carve((size_t)N * 4);     // deg then d^{-1/2}
    float* bufA  = (float*)carve((size_t)N * 256 * 4);
    float* bufB  = (float*)carve((size_t)N * 256 * 4);
    float* bufC  = (float*)carve((size_t)N * 256 * 4);
    float* x1s   = (float*)carve((size_t)N * 256 * 4);
    float* x2s   = (float*)carve((size_t)N * 128 * 4);
    float* gl    = (float*)carve((size_t)N * 128 * 4);
    float* gr    = (float*)carve((size_t)N * 128 * 4);
    float* elog  = (float*)carve((size_t)M * 4 * 4);
    float* denom = (float*)carve((size_t)N * 4 * 4);
    float* maxb  = (float*)carve((size_t)N * 4 * 4);

    auto blocks = [](long n) { return dim3((unsigned)((n + 255) / 256)); };
    auto fill = [&](float* p, float v, long n) {
        k_fill<<<blocks(n), 256, 0, stream>>>(p, v, n);
    };
    auto gemm = [&](const float* A, const float* B, float* C, int Kd, int Nd) {
        dim3 g((N + BM - 1) / BM, (Nd + BN - 1) / BN);
        gemm_wmma_bf16<<<g, 256, 0, stream>>>(A, B, C, N, Kd, Nd);
    };
    auto prop = [&](const float* in, float* o, int F) {
        fill(o, 0.f, (long)N * F);
        long tot = (long)M * (F >> 2);
        k_prop<<<blocks(tot), 256, 0, stream>>>(in, srcA, dstA, norm, o, M, F);
    };
    auto bias_relu = [&](const float* z, const float* b, float* o, int F) {
        long n = (long)N * F;
        k_bias_relu<<<blocks(n), 256, 0, stream>>>(z, b, o, n, F);
    };
    // full GCN layer: relu(prop(x@W) + b)
    auto gcn = [&](const float* xin, const float* W, const float* b,
                   float* o, int Kd, int Nd) {
        gemm(xin, W, bufA, Kd, Nd);
        prop(bufA, bufB, Nd);
        bias_relu(bufB, b, o, Nd);
    };

    // ---- graph preprocessing ----
    k_edges<<<blocks(M), 256, 0, stream>>>(ei, srcA, dstA, E, M);
    fill(dinv, 0.f, N);
    k_deg  <<<blocks(M), 256, 0, stream>>>(dstA, dinv, M);
    k_dinv <<<blocks(N), 256, 0, stream>>>(dinv, N);
    k_norm <<<blocks(M), 256, 0, stream>>>(srcA, dstA, dinv, norm, M);

    // ---- GCN stack ----
    gcn(x,    w_g1, b_g1, x1s,  256, 256);   // x1 [N,256]
    gcn(x1s,  w_g2, b_g2, x2s,  256, 128);   // x2 [N,128]
    gcn(x2s,  w_g3, b_g3, bufC, 128,  64);   // [N,64]
    gcn(bufC, w_g4, b_g4, bufC,  64,  32);   // [N,32]

    // ---- GATv2 (4 heads x 32) ----
    gemm(bufC, w_attl, gl, 32, 128);
    gemm(bufC, w_attr, gr, 32, 128);
    k_gat_logits<<<blocks((long)M * 4), 256, 0, stream>>>(gl, gr, att, srcA, dstA, elog, M);
    fill(maxb, -3.4e38f, (long)N * 4);
    k_gat_max<<<blocks((long)M * 4), 256, 0, stream>>>(elog, dstA, maxb, M);
    fill(denom, 0.f, (long)N * 4);
    k_gat_exp<<<blocks((long)M * 4), 256, 0, stream>>>(elog, dstA, maxb, denom, M);
    fill(bufB, 0.f, (long)N * 128);
    k_gat_aggr<<<blocks((long)M * 32), 256, 0, stream>>>(gl, elog, denom, srcA, dstA, bufB, M);
    bias_relu(bufB, b_att, bufC, 128);       // x3 [N,128]

    // ---- GCN2 #1 (128, x0 = x2) ----
    prop(bufC, bufA, 128);
    k_combine_alpha<<<blocks((long)N * 128), 256, 0, stream>>>(bufA, x2s, bufB, (long)N * 128);
    gemm(bufB, w_c1, bufA, 128, 128);
    k_gcn2_out<<<blocks((long)N * 128), 256, 0, stream>>>(bufB, bufA, bufC, (long)N * 128);

    // ---- GCN 128 -> 256 ----
    gcn(bufC, w_g5, b_g5, bufC, 128, 256);   // [N,256]

    // ---- GCN2 #2 (256, x0 = x1) ----
    prop(bufC, bufA, 256);
    k_combine_alpha<<<blocks((long)N * 256), 256, 0, stream>>>(bufA, x1s, bufB, (long)N * 256);
    gemm(bufB, w_c2, bufA, 256, 256);
    k_gcn2_out<<<blocks((long)N * 256), 256, 0, stream>>>(bufB, bufA, bufC, (long)N * 256);

    // ---- classifier GCN 256 -> 16 + softmax ----
    gemm(bufC, w_cls, bufA, 256, 16);
    prop(bufA, bufB, 16);
    k_softmax16<<<blocks(N), 256, 0, stream>>>(bufB, b_cls, out, N);
}
